// simOTA_20701742367352
// MI455X (gfx1250) — compile-verified, matrix operand order checked by reading, not run on hardware
//
#include <hip/hip_runtime.h>
#include <hip/hip_bf16.h>
#include <stdint.h>

#define GMAX       128
#define TOPK       10
#define SPLIT      8      // blocks per GT row in pass 1a (occupancy: G*SPLIT = 1024 blocks)
#define P1_THREADS 256
#define P1B_THREADS 128
#define P2_THREADS 256
#define P2_ITEMS   8
#define IOU_W      3.0f
#define EPSV       1e-8f

typedef __attribute__((ext_vector_type(16))) _Float16 v16h;
typedef __attribute__((ext_vector_type(8)))  float    v8f;

static __device__ __forceinline__ float iou_pair(float gx1, float gy1, float gx2, float gy2,
                                                 float px1, float py1, float px2, float py2) {
    float areaG = (gx2 - gx1) * (gy2 - gy1);
    float areaP = (px2 - px1) * (py2 - py1);
    float iw = fmaxf(fminf(gx2, px2) - fmaxf(gx1, px1), 0.0f);
    float ih = fmaxf(fminf(gy2, py2) - fmaxf(gy1, py1), 0.0f);
    float inter = iw * ih;
    float uni = areaG + areaP - inter;
    return inter > 0.0f ? inter / uni : 0.0f;
}

static __device__ __forceinline__ float softplus_neg(float x) {
    // -log(sigmoid(x)) = softplus(-x), numerically stable
    return x > 0.0f ? log1pf(__expf(-x)) : (-x + log1pf(__expf(x)));
}

// merge two sorted-descending 10-lists, keep top-10 in a
static __device__ __forceinline__ void merge_desc(float* a, const float* b) {
    float out[TOPK];
    int ia = 0, ib = 0;
#pragma unroll
    for (int i = 0; i < TOPK; ++i) {
        float av = a[ia], bv = b[ib];
        if (av >= bv) { out[i] = av; ++ia; } else { out[i] = bv; ++ib; }
    }
#pragma unroll
    for (int i = 0; i < TOPK; ++i) a[i] = out[i];
}

// merge two sorted-ascending 10-lists, keep lowest-10 in a
static __device__ __forceinline__ void merge_asc(float* a, const float* b) {
    float out[TOPK];
    int ia = 0, ib = 0;
#pragma unroll
    for (int i = 0; i < TOPK; ++i) {
        float av = a[ia], bv = b[ib];
        if (av <= bv) { out[i] = av; ++ia; } else { out[i] = bv; ++ib; }
    }
#pragma unroll
    for (int i = 0; i < TOPK; ++i) a[i] = out[i];
}

// Pass 1a: G*SPLIT blocks. Block (g,p) scans its N/SPLIT anchor slice for GT g,
// keeps per-thread top-10 (iou desc, cost asc), LDS tree-merges to a block-level
// partial list, writes it to workspace.
__global__ void __launch_bounds__(P1_THREADS)
simota_pass1a_partial(const float* __restrict__ prop, const float* __restrict__ logits,
                      const float* __restrict__ gtb,
                      float* __restrict__ partIou, float* __restrict__ partCst, int N) {
    __shared__ float sIou[P1_THREADS * TOPK];
    __shared__ float sCst[P1_THREADS * TOPK];
    const int bid = blockIdx.x;
    const int g = bid / SPLIT;
    const int p = bid % SPLIT;
    const int tid = threadIdx.x;

    const int slice = (N + SPLIT - 1) / SPLIT;
    const int start = p * slice;
    const int end = min(start + slice, N);

    const float gx1 = gtb[g * 4 + 0], gy1 = gtb[g * 4 + 1];
    const float gx2 = gtb[g * 4 + 2], gy2 = gtb[g * 4 + 3];

    float ti[TOPK], sc[TOPK];
#pragma unroll
    for (int i = 0; i < TOPK; ++i) { ti[i] = -1.0f; sc[i] = 3.4e38f; }

    for (int n = start + tid; n < end; n += P1_THREADS) {
        const float4 pb = ((const float4*)prop)[n];
        float iou = iou_pair(gx1, gy1, gx2, gy2, pb.x, pb.y, pb.z, pb.w);
        float cost = softplus_neg(logits[n]) - IOU_W * __logf(iou + EPSV);
        if (iou > ti[TOPK - 1]) {
            float v = iou;
#pragma unroll
            for (int i = 0; i < TOPK; ++i) {
                float hi = fmaxf(v, ti[i]), lo = fminf(v, ti[i]);
                ti[i] = hi; v = lo;
            }
        }
        if (cost < sc[TOPK - 1]) {
            float v = cost;
#pragma unroll
            for (int i = 0; i < TOPK; ++i) {
                float lo = fminf(v, sc[i]), hi = fmaxf(v, sc[i]);
                sc[i] = lo; v = hi;
            }
        }
    }
#pragma unroll
    for (int i = 0; i < TOPK; ++i) {
        sIou[tid * TOPK + i] = ti[i];
        sCst[tid * TOPK + i] = sc[i];
    }
    __syncthreads();
    for (int s = P1_THREADS / 2; s > 0; s >>= 1) {
        if (tid < s) {
            merge_desc(&sIou[tid * TOPK], &sIou[(tid + s) * TOPK]);
            merge_asc(&sCst[tid * TOPK], &sCst[(tid + s) * TOPK]);
        }
        __syncthreads();
    }
    if (tid < TOPK) {
        partIou[bid * TOPK + tid] = sIou[tid];
        partCst[bid * TOPK + tid] = sCst[tid];
    }
}

// Pass 1b: G blocks. Tree-merge the SPLIT partial lists of each GT row,
// compute dynamic_k and the k-th-smallest-cost threshold, emit the 8-float
// GT record {x1,y1,x2,y2, threshold, class_bits, 0, 0}.
__global__ void __launch_bounds__(P1B_THREADS)
simota_pass1b_merge(const float* __restrict__ gtb, const int* __restrict__ gtc,
                    const float* __restrict__ partIou, const float* __restrict__ partCst,
                    float* __restrict__ gtrec) {
    __shared__ float mIou[SPLIT * TOPK];
    __shared__ float mCst[SPLIT * TOPK];
    const int g = blockIdx.x;
    const int tid = threadIdx.x;

    if (tid < SPLIT * TOPK) {
        mIou[tid] = partIou[g * SPLIT * TOPK + tid];
        mCst[tid] = partCst[g * SPLIT * TOPK + tid];
    }
    __syncthreads();
    for (int s = SPLIT / 2; s > 0; s >>= 1) {
        if (tid < s) {
            merge_desc(&mIou[tid * TOPK], &mIou[(tid + s) * TOPK]);
            merge_asc(&mCst[tid * TOPK], &mCst[(tid + s) * TOPK]);
        }
        __syncthreads();
    }
    if (tid == 0) {
        float sum = 0.0f;
#pragma unroll
        for (int i = 0; i < TOPK; ++i) sum += mIou[i];
        int k = (int)sum;             // truncation, like .astype(int32)
        if (k < 1) k = 1;
        if (k > TOPK) k = TOPK;
        gtrec[g * 8 + 0] = gtb[g * 4 + 0];
        gtrec[g * 8 + 1] = gtb[g * 4 + 1];
        gtrec[g * 8 + 2] = gtb[g * 4 + 2];
        gtrec[g * 8 + 3] = gtb[g * 4 + 3];
        gtrec[g * 8 + 4] = mCst[k - 1];        // k-th smallest cost = row threshold
        gtrec[g * 8 + 5] = __int_as_float(gtc[g]);
        gtrec[g * 8 + 6] = 0.0f;
        gtrec[g * 8 + 7] = 0.0f;
    }
}

// Pass 2: per-anchor matching + conflict resolution. GT table staged to LDS via
// the CDNA5 async global->LDS path (ASYNCcnt); the inner loop reads each GT
// record as two ds_load_b128; num_fg is folded per wave on the matrix pipe.
__global__ void __launch_bounds__(P2_THREADS)
simota_pass2_match(const float* __restrict__ prop, const float* __restrict__ logits,
                   const float* __restrict__ gtrec,
                   float* __restrict__ out_fg, float* __restrict__ out_mg,
                   float* __restrict__ out_pi, float* __restrict__ out_cls,
                   float* __restrict__ out_numfg, int N, int G) {
    __shared__ __align__(16) float srec[GMAX * 8];  // 4 KB GT table
    const int tid = threadIdx.x;

    // Async-stage G*8 floats (G<=128 -> up to 1024 dwords) into LDS.
    {
        uint32_t lds_base = (uint32_t)(uintptr_t)(&srec[0]);
        uint64_t gbase = (uint64_t)(uintptr_t)gtrec;
        const int total = G * 8;
#pragma unroll
        for (int r = 0; r < 4; ++r) {
            int idx = tid + r * P2_THREADS;
            if (idx < total) {
                uint32_t lds_addr = lds_base + (uint32_t)idx * 4u;
                uint32_t goff = (uint32_t)idx * 4u;
                asm volatile("global_load_async_to_lds_b32 %0, %1, %2"
                             :: "v"(lds_addr), "v"(goff), "s"(gbase) : "memory");
            }
        }
        asm volatile("s_wait_asynccnt 0x0" ::: "memory");
    }
    __syncthreads();

    const float4* srec4 = (const float4*)srec;
    const int base = blockIdx.x * (P2_THREADS * P2_ITEMS);
    int cnt_total = 0;

#pragma unroll
    for (int j = 0; j < P2_ITEMS; ++j) {
        int n = base + j * P2_THREADS + tid;
        const float4 pb = ((const float4*)prop)[n];
        float obj = softplus_neg(logits[n]);

        int cnt = 0, firstG = 0, minG = 0;
        float iouFirst = 0.0f, minCost = 3.4e38f, minIou = 0.0f;
        for (int g = 0; g < G; ++g) {
            float4 r0 = srec4[g * 2];       // box
            float4 r1 = srec4[g * 2 + 1];   // {threshold, class_bits, 0, 0}
            float iou = iou_pair(r0.x, r0.y, r0.z, r0.w, pb.x, pb.y, pb.z, pb.w);
            float cost = obj - IOU_W * __logf(iou + EPSV);
            if (cost < minCost) { minCost = cost; minG = g; minIou = iou; }  // first-occurrence argmin
            if (cost <= r1.x) {
                if (cnt == 0) { firstG = g; iouFirst = iou; }
                ++cnt;
            }
        }
        float fg, mg, pi, cls;
        if (cnt == 0) {
            fg = 0.0f; mg = 0.0f; pi = 0.0f; cls = -1.0f;
        } else if (cnt == 1) {
            fg = 1.0f; mg = (float)firstG; pi = iouFirst;
            cls = (float)__float_as_int(srec[firstG * 8 + 5]);
        } else {  // conflict: whole column replaced by one-hot at the column argmin
            fg = 1.0f; mg = (float)minG; pi = minIou;
            cls = (float)__float_as_int(srec[minG * 8 + 5]);
        }
        out_fg[n] = fg;
        out_mg[n] = mg;
        out_pi[n] = pi;
        out_cls[n] = cls;
        cnt_total += (cnt > 0) ? 1 : 0;
    }

    // num_fg: fold the 32 per-lane counts of this wave on the matrix pipe.
    // A[m][k] has only K=0 (lanes 0-15) / K=8 (lanes 16-31) set to count;
    // B = ones -> D[m][n] = cnt(lane m) + cnt(lane m+16).
    {
        v16h a, b;
#pragma unroll
        for (int i = 0; i < 16; ++i) { a[i] = (_Float16)0.0f; b[i] = (_Float16)1.0f; }
        a[0] = (_Float16)(float)cnt_total;   // exact: cnt_total <= 8
        v8f c = {};
        c = __builtin_amdgcn_wmma_f32_16x16x32_f16(
                /*neg_a=*/false, a, /*neg_b=*/false, b,
                /*c_mod=*/(short)0, c, /*reuse_a=*/false, /*reuse_b=*/false);
        float s = 0.0f;
#pragma unroll
        for (int r = 0; r < 8; ++r) s += c[r];          // 8 rows of my column
        s += __shfl_xor(s, 16, 32);                      // + the other 8 rows
        if ((tid & 31) == 0) atomicAdd(out_numfg, s);    // one atomic per wave
    }
}

extern "C" void kernel_launch(void* const* d_in, const int* in_sizes, int n_in,
                              void* d_out, int out_size, void* d_ws, size_t ws_size,
                              hipStream_t stream) {
    const float* prop   = (const float*)d_in[0];   // [N,4]
    const float* logits = (const float*)d_in[1];   // [N]
    const float* gtb    = (const float*)d_in[2];   // [G,4]
    const int*   gtc    = (const int*)d_in[3];     // [G]
    const int N = in_sizes[1];
    const int G = in_sizes[3];

    float* out = (float*)d_out;      // [fg | matched_gt | pred_iou | classes | num_fg]

    // workspace layout
    float* partIou = (float*)d_ws;                         // G*SPLIT*TOPK
    float* partCst = partIou + (size_t)G * SPLIT * TOPK;   // G*SPLIT*TOPK
    float* gtrec   = partCst + (size_t)G * SPLIT * TOPK;   // G*8

    hipMemsetAsync(out + 4 * (size_t)N, 0, sizeof(float), stream);  // zero num_fg (capturable)

    simota_pass1a_partial<<<G * SPLIT, P1_THREADS, 0, stream>>>(
        prop, logits, gtb, partIou, partCst, N);

    simota_pass1b_merge<<<G, P1B_THREADS, 0, stream>>>(
        gtb, gtc, partIou, partCst, gtrec);

    int blocks = (N + P2_THREADS * P2_ITEMS - 1) / (P2_THREADS * P2_ITEMS);
    simota_pass2_match<<<blocks, P2_THREADS, 0, stream>>>(
        prop, logits, gtrec,
        out, out + (size_t)N, out + 2 * (size_t)N, out + 3 * (size_t)N,
        out + 4 * (size_t)N, N, G);
}
